// GAT_16561393893826
// MI455X (gfx1250) — matrix operand order repaired and checked by reading, not compiled, not run on hardware
//
#include <hip/hip_runtime.h>
#include <hip/hip_fp16.h>

#define B_ 4
#define N_ 2048
#define D_ 128
#define H_ 256

typedef __attribute__((ext_vector_type(16))) _Float16 v16h;
typedef __attribute__((ext_vector_type(8)))  _Float16 v8h;
typedef __attribute__((ext_vector_type(4)))  _Float16 v4h;
typedef __attribute__((ext_vector_type(8)))  float    v8f;
typedef __attribute__((ext_vector_type(4)))  float    f4;
typedef __attribute__((ext_vector_type(4)))  int      i4;
typedef __attribute__((ext_vector_type(4)))  unsigned int u32x4;
typedef __attribute__((ext_vector_type(4)))  int      i32x4v;
typedef __attribute__((ext_vector_type(8)))  int      i32x8v;

#if defined(__has_builtin)
#if __has_builtin(__builtin_amdgcn_tensor_load_to_lds)
#define HAVE_TDM 1
#else
#define HAVE_TDM 0
#endif
#else
#define HAVE_TDM 0
#endif

static __device__ __forceinline__ v8f wmma16x16x32(v16h a, v16h b, v8f c) {
  // (neg_a, A, neg_b, B, c_mod, C, reuse_a, reuse_b)
  return __builtin_amdgcn_wmma_f32_16x16x32_f16(false, a, false, b, (short)0, c,
                                                false, false);
}

#if HAVE_TDM
// DMA a [128 rows x 256 cols] f16 tile of hT (row stride N_ elements) from
// global memory into LDS at byte offset lds_off. 2D descriptor; groups 2/3
// zero (tile_dim2 = 0 -> unused). Tracked by TENSORcnt.
static __device__ __forceinline__ void tdm_load_tile(const _Float16* gptr,
                                                     unsigned lds_off) {
  const unsigned long long ga = (unsigned long long)(size_t)gptr;
  u32x4 g0;
  g0[0] = 1u;                                   // count=1, user descriptor
  g0[1] = lds_off;                              // LDS byte address
  g0[2] = (unsigned)ga;                         // global_addr[31:0]
  g0[3] = (unsigned)((ga >> 32) & 0x01FFFFFFull) | 0x80000000u;  // type=2
  i32x8v g1;
  g1[0] = 1 << 16;             // data_size = 1 -> 2 bytes/elem
  g1[1] = (int)(2048u << 16);  // tensor_dim0 = N_ (low16 in dw1[31:16])
  g1[2] = (int)(128u << 16);   // tensor_dim0 hi = 0 | tensor_dim1 = D_
  g1[3] = (int)(256u << 16);   // tensor_dim1 hi = 0 | tile_dim0 = 256
  g1[4] = 128;                 // tile_dim1 = 128, tile_dim2 = 0
  g1[5] = 2048;                // tensor_dim0_stride = N_ (elements)
  g1[6] = 0;
  g1[7] = 0;
  const i32x4v z4 = {0, 0, 0, 0};
#if __clang_major__ >= 23
  const i32x8v z8 = {0, 0, 0, 0, 0, 0, 0, 0};
  __builtin_amdgcn_tensor_load_to_lds(g0, g1, z4, z4, z8, 0);
#else
  __builtin_amdgcn_tensor_load_to_lds(g0, g1, z4, z4, 0);
#endif
}
#endif

// ---------------------------------------------------------------------------
// f32 -> f16 elementwise convert (n must be a multiple of 4).
// ---------------------------------------------------------------------------
__global__ __launch_bounds__(256) void k_cvt(const float* __restrict__ src,
                                             _Float16* __restrict__ dst,
                                             int n) {
  const int i = (blockIdx.x * 256 + threadIdx.x) * 4;
  if (i < n) {
    const f4 v = *(const f4*)(src + i);
    v4h o;
    o[0] = (_Float16)v[0];
    o[1] = (_Float16)v[1];
    o[2] = (_Float16)v[2];
    o[3] = (_Float16)v[3];
    *(v4h*)(dst + i) = o;
  }
}

// ---------------------------------------------------------------------------
// h = relu(xh @ Wf^T + b); all-f16 operands. Stores h [B,N,D] and hT [B,D,N].
// grid = B*N/16 blocks, 256 threads (8 waves; wave w -> e-tile w*16).
// ---------------------------------------------------------------------------
__global__ __launch_bounds__(256) void k_linear(
    const _Float16* __restrict__ xh, const _Float16* __restrict__ Wf,
    const float* __restrict__ bias, _Float16* __restrict__ h,
    _Float16* __restrict__ hT) {
  const int wave = threadIdx.x >> 5;
  const int lane = threadIdx.x & 31;
  const int lm = lane & 15;
  const int hi = lane >> 4;
  const int b  = blockIdx.x / (N_ / 16);
  const int n0 = (blockIdx.x % (N_ / 16)) * 16;
  const int e0 = wave * 16;

  const _Float16* xrow = xh + ((size_t)b * N_ + (n0 + lm)) * D_;  // A: lane=row
  const _Float16* wrow = Wf + (size_t)(e0 + lm) * D_;             // B: lane=col

  v8f c = {};
  #pragma unroll
  for (int kb = 0; kb < D_; kb += 32) {
    const int abase = kb + hi * 16;  // A: 16 contiguous K per lane
    const v8h p0 = *(const v8h*)(xrow + abase);
    const v8h p1 = *(const v8h*)(xrow + abase + 8);
    const v16h a = __builtin_shufflevector(p0, p1, 0, 1, 2, 3, 4, 5, 6, 7, 8, 9,
                                           10, 11, 12, 13, 14, 15);
    const int bbase = kb + hi * 16;  // B: 16 contiguous K per lane
    const v16h bm = *(const v16h*)(wrow + bbase);
    c = wmma16x16x32(a, bm, c);
  }
  const float bv = bias[e0 + lm];    // C: lane -> column e0+lm
  v8h hv;
  #pragma unroll
  for (int r = 0; r < 8; ++r) {
    const float v = c[r] + bv;
    hv[r] = (_Float16)(v > 0.f ? v : 0.f);
  }
  // hT[b, e0+lm, n0+hi*8 .. +7] contiguous in r
  *(v8h*)(hT + ((size_t)b * D_ + (e0 + lm)) * N_ + n0 + hi * 8) = hv;
  #pragma unroll
  for (int r = 0; r < 8; ++r) {
    const int nrow = n0 + hi * 8 + r;
    h[((size_t)b * N_ + nrow) * D_ + e0 + lm] = hv[r];
  }
}

// ---------------------------------------------------------------------------
// f1 = h @ a[:D], f2 = h @ a[D:].  One thread per row.
// ---------------------------------------------------------------------------
__global__ __launch_bounds__(256) void k_f12(
    const _Float16* __restrict__ h, const float* __restrict__ a,
    float* __restrict__ f1, float* __restrict__ f2) {
  const int idx = blockIdx.x * 256 + threadIdx.x;  // b*N + i
  const _Float16* hr = h + (size_t)idx * D_;
  float s1 = 0.f, s2 = 0.f;
  #pragma unroll 8
  for (int d = 0; d < D_; ++d) {
    const float hv = (float)hr[d];
    s1 += hv * a[d];
    s2 += hv * a[D_ + d];
  }
  f1[idx] = s1;
  f2[idx] = s2;
}

// ---------------------------------------------------------------------------
// Per-row masked softmax stats: rowmax and 1/sum(exp). One wave per row.
// grid = B*N/8, 256 threads (8 waves).
// ---------------------------------------------------------------------------
__global__ __launch_bounds__(256) void k_stats(
    const int* __restrict__ adj, const float* __restrict__ f1,
    const float* __restrict__ f2, float* __restrict__ rowmax,
    float* __restrict__ rdenom) {
  const int lane = threadIdx.x & 31;
  const int idx = blockIdx.x * 8 + (threadIdx.x >> 5);  // b*N + i
  const int b = idx / N_;
  const int* arow = adj + (size_t)idx * N_;
  const float* f2b = f2 + (size_t)b * N_;
  const float fi = f1[idx];

  float m = -3.0e38f, s = 0.f;
  for (int j = lane; j < N_; j += 32) {
    __builtin_prefetch(arow + j + 512, 0, 1);  // global_prefetch_b8, L2 warm
    if (arow[j] > 0) {
      const float v = fi + f2b[j];
      const float e = v > 0.f ? v : 0.01f * v;  // leaky_relu
      const float mn = fmaxf(m, e);
      s = s * __expf(m - mn) + __expf(e - mn);
      m = mn;
    }
  }
  #pragma unroll
  for (int off = 16; off > 0; off >>= 1) {
    const float mo = __shfl_xor(m, off, 32);
    const float so = __shfl_xor(s, off, 32);
    const float mn = fmaxf(m, mo);
    s = s * __expf(m - mn) + so * __expf(mo - mn);
    m = mn;
  }
  if (lane == 0) {
    rowmax[idx] = m;
    rdenom[idx] = (s > 0.f) ? 1.f / s : 0.f;
  }
}

// ---------------------------------------------------------------------------
// xout = x + softmax(mask(e)) @ h  (also emits f16 copy for next layer GEMM).
// grid = B*N/16, 256 threads (8 waves). Wave w owns d-tile w. Waves build 8
// att tiles (16x32 f16, A per-lane layout) in LDS per outer step; hT slabs
// are double-buffered into LDS by the Tensor Data Mover (TENSORcnt).
// Dynamic LDS: 2*64KB slabs + 8KB att = 139264 bytes.
// ---------------------------------------------------------------------------
__global__ __launch_bounds__(256) void k_attn(
    const float* __restrict__ x, const int* __restrict__ adj,
    const _Float16* __restrict__ hT, const float* __restrict__ f1,
    const float* __restrict__ f2, const float* __restrict__ rowmax,
    const float* __restrict__ rdenom, float* __restrict__ xout,
    _Float16* __restrict__ xh_out) {
  extern __shared__ __align__(32) char smem[];
  _Float16* buf  = (_Float16*)smem;                       // [2][128][256]
  _Float16* satt = (_Float16*)(smem + 2 * 128 * 256 * 2); // [8][32][16]

  const int wave = threadIdx.x >> 5;
  const int lane = threadIdx.x & 31;
  const int lm = lane & 15;
  const int hi = lane >> 4;
  const int b  = blockIdx.x / (N_ / 16);
  const int n0 = (blockIdx.x % (N_ / 16)) * 16;

  const size_t ibase = (size_t)b * N_ + (n0 + lm);  // row handled by this lane
  const float fi = f1[ibase];
  const float mi = rowmax[ibase];
  const float rd = rdenom[ibase];
  const int* arow = adj + ibase * N_;
  const float* f2b = f2 + (size_t)b * N_;
  const int d0 = wave * 16;

#if HAVE_TDM
  const _Float16* gbase = hT + (size_t)b * D_ * N_;
  const unsigned lds0 = (unsigned)(size_t)buf;
  if (wave == 0) tdm_load_tile(gbase, lds0);  // pre-issue slab for jb=0
#else
  const _Float16* hTrow = hT + ((size_t)b * D_ + (d0 + lm)) * N_;
#endif

  v8f acc = {};
  for (int jb = 0; jb < N_; jb += 256) {
    const int cur = (jb >> 8) & 1;
    // --- build att tile for j-tile (jb + wave*32), A per-lane layout ---
    const int j0 = jb + wave * 32;
    _Float16 av[16];
    #pragma unroll
    for (int half = 0; half < 2; ++half) {
      const int kb = j0 + hi * 16 + half * 8;
      const i4 a0 = *(const i4*)(arow + kb);
      const i4 a1 = *(const i4*)(arow + kb + 4);
      const f4 g0 = *(const f4*)(f2b + kb);
      const f4 g1 = *(const f4*)(f2b + kb + 4);
      #pragma unroll
      for (int t = 0; t < 4; ++t) {
        const float v = fi + g0[t];
        const float e = v > 0.f ? v : 0.01f * v;
        av[half * 8 + t] = (_Float16)((a0[t] > 0) ? __expf(e - mi) * rd : 0.f);
      }
      #pragma unroll
      for (int t = 0; t < 4; ++t) {
        const float v = fi + g1[t];
        const float e = v > 0.f ? v : 0.01f * v;
        av[half * 8 + 4 + t] =
            (_Float16)((a1[t] > 0) ? __expf(e - mi) * rd : 0.f);
      }
    }
    #pragma unroll
    for (int t = 0; t < 16; ++t) satt[((size_t)wave * 32 + lane) * 16 + t] = av[t];

#if HAVE_TDM
    if (wave == 0) __builtin_amdgcn_s_wait_tensorcnt(0);  // slab cur arrived
#endif
    __syncthreads();  // satt visible; slab cur ready; old slab reads done
#if HAVE_TDM
    if (wave == 0 && jb + 256 < N_)  // prefetch next slab into other buffer
      tdm_load_tile(gbase + (size_t)(jb + 256), lds0 + (cur ^ 1) * 65536u);
#endif

    // --- 8 WMMAs against hT B-tiles ---
    #pragma unroll
    for (int t = 0; t < 8; ++t) {
      const v16h a = *(const v16h*)(satt + ((size_t)t * 32 + lane) * 16);
#if HAVE_TDM
      const _Float16* src =
          buf + (size_t)cur * (128 * 256) + (d0 + lm) * 256 + t * 32 + hi * 16;
      const v16h bm = *(const v16h*)src;
#else
      const _Float16* srcg = hTrow + (jb + t * 32) + hi * 16;
      const v8h s0 = *(const v8h*)(srcg);
      const v8h s1 = *(const v8h*)(srcg + 8);
      const v16h bm = __builtin_shufflevector(s0, s1, 0, 1, 2, 3, 4, 5, 6, 7, 8,
                                              9, 10, 11, 12, 13, 14, 15);
#endif
      acc = wmma16x16x32(a, bm, acc);
    }
    __syncthreads();
  }

  // residual add + store f32 and f16 (C: row = n0 + r + hi*8, col = d0 + lm)
  #pragma unroll
  for (int r = 0; r < 8; ++r) {
    const int nrow = n0 + r + hi * 8;
    const size_t off = ((size_t)b * N_ + nrow) * D_ + d0 + lm;
    const float nv = x[off] + acc[r];
    xout[off] = nv;
    xh_out[off] = (_Float16)nv;
  }
}

// ---------------------------------------------------------------------------
// out = relu(xh @ Wt^T + bt), H=256; f16 operands. grid = B*N/16, 512 threads.
// ---------------------------------------------------------------------------
__global__ __launch_bounds__(512) void k_final(
    const _Float16* __restrict__ xh, const _Float16* __restrict__ Wtf,
    const float* __restrict__ bt, float* __restrict__ out) {
  const int wave = threadIdx.x >> 5;  // 0..15 -> e-tile
  const int lane = threadIdx.x & 31;
  const int lm = lane & 15;
  const int hi = lane >> 4;
  const int b  = blockIdx.x / (N_ / 16);
  const int n0 = (blockIdx.x % (N_ / 16)) * 16;
  const int e0 = wave * 16;

  const _Float16* xrow = xh + ((size_t)b * N_ + (n0 + lm)) * D_;
  const _Float16* wrow = Wtf + (size_t)(e0 + lm) * D_;

  v8f c = {};
  #pragma unroll
  for (int kb = 0; kb < D_; kb += 32) {
    const int abase = kb + hi * 16;
    const v8h p0 = *(const v8h*)(xrow + abase);
    const v8h p1 = *(const v8h*)(xrow + abase + 8);
    const v16h a = __builtin_shufflevector(p0, p1, 0, 1, 2, 3, 4, 5, 6, 7, 8, 9,
                                           10, 11, 12, 13, 14, 15);
    const int bbase = kb + hi * 16;
    const v16h bm = *(const v16h*)(wrow + bbase);
    c = wmma16x16x32(a, bm, c);
  }
  const float bv = bt[e0 + lm];
  #pragma unroll
  for (int r = 0; r < 8; ++r) {
    const int nrow = n0 + r + hi * 8;
    const float v = c[r] + bv;
    out[((size_t)b * N_ + nrow) * H_ + e0 + lm] = v > 0.f ? v : 0.f;
  }
}

// ---------------------------------------------------------------------------
// Host-side orchestration. Inputs (setup_inputs order):
// 0:x 1:adj 2:W0 3:b0 4:a0 5:W1 6:b1 7:a1 8:W2 9:b2 10:a2 11:Wt 12:bt
// ---------------------------------------------------------------------------
extern "C" void kernel_launch(void* const* d_in, const int* in_sizes, int n_in,
                              void* d_out, int out_size, void* d_ws,
                              size_t ws_size, hipStream_t stream) {
  const float* x   = (const float*)d_in[0];
  const int*   adj = (const int*)d_in[1];
  const float* Ws[3] = {(const float*)d_in[2], (const float*)d_in[5],
                        (const float*)d_in[8]};
  const float* bs[3] = {(const float*)d_in[3], (const float*)d_in[6],
                        (const float*)d_in[9]};
  const float* as[3] = {(const float*)d_in[4], (const float*)d_in[7],
                        (const float*)d_in[10]};
  const float* Wt = (const float*)d_in[11];
  const float* bt = (const float*)d_in[12];

  const size_t MB = 1024 * 1024;
  char* ws = (char*)d_ws;
  float*    xa   = (float*)(ws);             // 4 MB  f32 x ping
  float*    xb   = (float*)(ws + 4 * MB);    // 4 MB  f32 x pong
  _Float16* h    = (_Float16*)(ws + 8 * MB);   // 2 MB
  _Float16* hT   = (_Float16*)(ws + 10 * MB);  // 2 MB
  _Float16* xh   = (_Float16*)(ws + 12 * MB);  // 2 MB  f16 layer input
  _Float16* wf0  = (_Float16*)(ws + 14 * MB);  // 32 KB each
  _Float16* wf1  = wf0 + D_ * D_;
  _Float16* wf2  = wf1 + D_ * D_;
  _Float16* wtf  = wf2 + D_ * D_;              // 64 KB
  float*    f1   = (float*)(wtf + H_ * D_);
  float*    f2   = f1 + B_ * N_;
  float*    rmax = f2 + B_ * N_;
  float*    rden = rmax + B_ * N_;
  _Float16* wfs[3] = {wf0, wf1, wf2};

  // One-time f16 conversions (weights + initial x).
  for (int l = 0; l < 3; ++l)
    k_cvt<<<dim3(D_ * D_ / 1024), dim3(256), 0, stream>>>(Ws[l], wfs[l],
                                                          D_ * D_);
  k_cvt<<<dim3(H_ * D_ / 1024), dim3(256), 0, stream>>>(Wt, wtf, H_ * D_);
  k_cvt<<<dim3(B_ * N_ * D_ / 1024), dim3(256), 0, stream>>>(x, xh,
                                                             B_ * N_ * D_);

  const size_t attn_lds = 2 * 128 * 256 * 2 + 8 * 32 * 16 * 2;  // 139264 B

  const float* xin = x;  // f32 residual source
  float* xout = xa;
  for (int l = 0; l < 3; ++l) {
    k_linear<<<dim3(B_ * N_ / 16), dim3(256), 0, stream>>>(xh, wfs[l], bs[l], h,
                                                           hT);
    k_f12<<<dim3(B_ * N_ / 256), dim3(256), 0, stream>>>(h, as[l], f1, f2);
    k_stats<<<dim3(B_ * N_ / 8), dim3(256), 0, stream>>>(adj, f1, f2, rmax,
                                                         rden);
    // writes f32 xout and refreshes xh (f16) for the next GEMM
    k_attn<<<dim3(B_ * N_ / 16), dim3(256), attn_lds, stream>>>(
        xin, adj, hT, f1, f2, rmax, rden, xout, xh);
    xin = xout;
    xout = (xout == xa) ? xb : xa;
  }
  k_final<<<dim3(B_ * N_ / 16), dim3(512), 0, stream>>>(xh, wtf, bt,
                                                        (float*)d_out);
}
